// MoEBlock_75290776699504
// MI455X (gfx1250) — matrix operand order repaired and checked by reading, not compiled, not run on hardware
//
#include <hip/hip_runtime.h>
#include <hip/hip_bf16.h>
#include <math.h>

typedef __bf16 bf16_t;
typedef __attribute__((ext_vector_type(8)))  bf16_t v8bf;
typedef __attribute__((ext_vector_type(16))) bf16_t v16bf;
typedef __attribute__((ext_vector_type(8)))  float  v8f;

constexpr int kB = 2, kSeq = 1024, kD = 1024, kH = 16, kHd = 64, kE = 8, kF = 4096;
constexpr int kT = kB * kSeq;                       // 2048 tokens
constexpr int kPadRows = kT * 2 + kE * 128;         // 5120 padded token-expert slots
constexpr float kEps = 1e-5f;
constexpr float kScale = 0.125f;                    // 1/sqrt(64)

constexpr int BM = 128, BN = 128, BK = 32, LDT = 40; // LDS K-stride (80B, 16B aligned)

// ---------------- CDNA5 async global->LDS copy (ASYNCcnt path) ----------------
__device__ inline void async_b128(const void* gsrc, void* ldst) {
  unsigned lds = (unsigned)(unsigned long long)(uintptr_t)ldst;
  unsigned long long ga = (unsigned long long)(uintptr_t)gsrc;
  asm volatile("global_load_async_to_lds_b128 %0, %1, off"
               :: "v"(lds), "v"(ga) : "memory");
}
__device__ inline void wait_async0() {
  asm volatile("s_wait_asynccnt 0x0" ::: "memory");
}

// ---------------- WMMA fragment helpers (LDS [row][k] layout) ----------------
// A 16x32 bf16 frag: lanes 0-15 hold K 0..7 & 16..23, lanes 16-31 hold K 8..15 & 24..31
__device__ inline v16bf frag_a(const bf16_t* rowp, int lhalf) {
  const int k0 = lhalf * 8;
  v8bf lo = *(const v8bf*)(rowp + k0);
  v8bf hi = *(const v8bf*)(rowp + k0 + 16);
  v16bf r;
#pragma unroll
  for (int i = 0; i < 8; ++i) { r[i] = lo[i]; r[i + 8] = hi[i]; }
  return r;
}
// B 32x16 bf16 frag (column-major staged as [n][k]): lanes 0-15 K 0..7 & 16..23, lanes 16-31 K 8..15 & 24..31
__device__ inline v16bf frag_b(const bf16_t* colp, int lhalf) {
  const int k0 = lhalf * 8;
  v8bf lo = *(const v8bf*)(colp + k0);
  v8bf hi = *(const v8bf*)(colp + k0 + 16);
  v16bf r;
#pragma unroll
  for (int i = 0; i < 8; ++i) { r[i] = lo[i]; r[i + 8] = hi[i]; }
  return r;
}
__device__ inline v8f wmma_bf16(v16bf a, v16bf b, v8f c) {
  return __builtin_amdgcn_wmma_f32_16x16x32_bf16(false, a, false, b, (short)0, c, false, false);
}
__device__ inline float gelu_tanh(float x) {
  return 0.5f * x * (1.0f + tanhf(0.7978845608028654f * (x + 0.044715f * x * x * x)));
}

// ---------------- LayerNorm (row per block), bf16 output ----------------
__global__ __launch_bounds__(256) void ln_kernel(const float* __restrict__ x,
                                                 const float* __restrict__ g,
                                                 const float* __restrict__ b,
                                                 bf16_t* __restrict__ out) {
  const int row = blockIdx.x;
  const float* xr = x + (size_t)row * kD;
  __shared__ float red[256];
  __shared__ float stat[2];
  const int tid = threadIdx.x;
  float s = 0.f;
  for (int c = tid; c < kD; c += 256) s += xr[c];
  red[tid] = s; __syncthreads();
  for (int off = 128; off > 0; off >>= 1) {
    if (tid < off) red[tid] += red[tid + off];
    __syncthreads();
  }
  if (tid == 0) stat[0] = red[0] / kD;
  __syncthreads();
  const float m = stat[0];
  float s2 = 0.f;
  for (int c = tid; c < kD; c += 256) { float d = xr[c] - m; s2 += d * d; }
  red[tid] = s2; __syncthreads();
  for (int off = 128; off > 0; off >>= 1) {
    if (tid < off) red[tid] += red[tid + off];
    __syncthreads();
  }
  if (tid == 0) stat[1] = rsqrtf(red[0] / kD + kEps);
  __syncthreads();
  const float rstd = stat[1];
  for (int c = tid; c < kD; c += 256)
    out[(size_t)row * kD + c] = (bf16_t)((xr[c] - m) * rstd * g[c] + b[c]);
}

// ---------------- Generic 128x128x32 WMMA GEMM, double-buffered pipeline ----------------
// C[M,N] f32 = A_bf16[M,K] * B_f32[K,N] (B converted bf16 while staging to LDS).
// Stage i+1's A-tile DMA and B-tile fp32 loads are issued before stage i's WMMAs,
// giving HBM weight streaming / WMMA overlap; one barrier per K-step.
enum { EPI_BF16 = 0, EPI_RESID = 1, EPI_GELU_BF16 = 2, EPI_MOE_SCATTER = 3 };

template<int MODE, bool GATHER, bool EXPERT>
__global__ __launch_bounds__(256) void gemm_kernel(
    const bf16_t* __restrict__ A, const float* __restrict__ Bw,
    const float* __restrict__ bias, void* __restrict__ Out,
    const float* __restrict__ resid, const float* __restrict__ lsv,
    const int* __restrict__ tok_list, const float* __restrict__ gate_list,
    const int* __restrict__ seg_off, float* __restrict__ moe_acc,
    const bf16_t* __restrict__ zsrc,
    int Nfull, int K, int lda) {
  __shared__ __align__(16) bf16_t As[2][BM][LDT];
  __shared__ __align__(16) bf16_t Bs[2][BN][LDT];   // transposed: [n][k]
  const int tid = threadIdx.x;
  const int lane = tid & 31, wave = tid >> 5;
  const int l16 = lane & 15, lhalf = lane >> 4;
  const int wm = wave & 3, wn = wave >> 2;          // 4 (M) x 2 (N) waves
  const int rowbase = blockIdx.y * BM;
  const int colbase = blockIdx.x * BN;

  int e = 0;
  if (EXPERT) {
    if (rowbase >= seg_off[kE]) return;             // uniform across block
    while (e < kE - 1 && seg_off[e + 1] <= rowbase) ++e;
  }
  const float* Bwp = EXPERT ? (Bw + (size_t)e * K * Nfull) : Bw;
  const float* bip = EXPERT ? (bias + (size_t)e * Nfull) : bias;

  // Per-thread staging coordinates
  const int a_r = (tid * 8) >> 5, a_c = (tid * 8) & 31;            // it 0
  const int a_r1 = ((tid + 256) * 8) >> 5, a_c1 = ((tid + 256) * 8) & 31;  // it 1

  // Per-thread A source pointers (resolve gather indices once; dead rows -> zero page)
  const bf16_t* arow[2];
  int astep[2];
  {
    int rr[2] = {a_r, a_r1}, cc[2] = {a_c, a_c1};
#pragma unroll
    for (int it = 0; it < 2; ++it) {
      if (GATHER) {
        int tk = tok_list[rowbase + rr[it]];
        if (tk >= 0) { arow[it] = A + (size_t)tk * lda + cc[it]; astep[it] = BK; }
        else         { arow[it] = zsrc;                           astep[it] = 0;  }
      } else {
        arow[it] = A + (size_t)(rowbase + rr[it]) * lda + cc[it];
        astep[it] = BK;
      }
    }
  }

  v8f acc[2][4] = {};
  float4 breg[4];

  // ---- prologue: stage k-block 0 into buffer 0 ----
  {
    async_b128(arow[0], &As[0][a_r][a_c]);   arow[0] += astep[0];
    async_b128(arow[1], &As[0][a_r1][a_c1]); arow[1] += astep[1];
#pragma unroll
    for (int it = 0; it < 4; ++it) {
      int idx = (tid + it * 256) * 4;
      int kk = idx >> 7, n = idx & 127;
      breg[it] = *(const float4*)(Bwp + (size_t)kk * Nfull + colbase + n);
    }
#pragma unroll
    for (int it = 0; it < 4; ++it) {
      int idx = (tid + it * 256) * 4;
      int kk = idx >> 7, n = idx & 127;
      Bs[0][n + 0][kk] = (bf16_t)breg[it].x;
      Bs[0][n + 1][kk] = (bf16_t)breg[it].y;
      Bs[0][n + 2][kk] = (bf16_t)breg[it].z;
      Bs[0][n + 3][kk] = (bf16_t)breg[it].w;
    }
    wait_async0();
    __syncthreads();
  }

  for (int kb = 0; kb < K; kb += BK) {
    const int cur = (kb / BK) & 1, nxt = cur ^ 1;
    const bool more = (kb + BK) < K;
    // issue next stage: async A DMA + fp32 B loads (kept in flight across the WMMAs)
    if (more) {
      async_b128(arow[0], &As[nxt][a_r][a_c]);   arow[0] += astep[0];
      async_b128(arow[1], &As[nxt][a_r1][a_c1]); arow[1] += astep[1];
#pragma unroll
      for (int it = 0; it < 4; ++it) {
        int idx = (tid + it * 256) * 4;
        int kk = idx >> 7, n = idx & 127;
        const float* bsrc = Bwp + (size_t)(kb + BK + kk) * Nfull + colbase + n;
        breg[it] = *(const float4*)bsrc;
        if (kb + 2 * BK < K) __builtin_prefetch(bsrc + (size_t)BK * Nfull, 0, 0);
      }
    }
    // compute current stage
    v16bf af[2], bfrag[4];
#pragma unroll
    for (int mi = 0; mi < 2; ++mi)
      af[mi] = frag_a(&As[cur][wm * 32 + mi * 16 + l16][0], lhalf);
#pragma unroll
    for (int ni = 0; ni < 4; ++ni)
      bfrag[ni] = frag_b(&Bs[cur][wn * 64 + ni * 16 + l16][0], lhalf);
#pragma unroll
    for (int mi = 0; mi < 2; ++mi)
#pragma unroll
      for (int ni = 0; ni < 4; ++ni)
        acc[mi][ni] = wmma_bf16(af[mi], bfrag[ni], acc[mi][ni]);
    // finish next stage: convert B to bf16 into alt buffer, then drain the DMA
    if (more) {
#pragma unroll
      for (int it = 0; it < 4; ++it) {
        int idx = (tid + it * 256) * 4;
        int kk = idx >> 7, n = idx & 127;
        Bs[nxt][n + 0][kk] = (bf16_t)breg[it].x;
        Bs[nxt][n + 1][kk] = (bf16_t)breg[it].y;
        Bs[nxt][n + 2][kk] = (bf16_t)breg[it].z;
        Bs[nxt][n + 3][kk] = (bf16_t)breg[it].w;
      }
      wait_async0();
    }
    __syncthreads();
  }

  // epilogue (C layout: lane holds col l16, rows j + 8*lhalf)
#pragma unroll
  for (int mi = 0; mi < 2; ++mi) {
#pragma unroll
    for (int ni = 0; ni < 4; ++ni) {
#pragma unroll
      for (int j = 0; j < 8; ++j) {
        const int grow = rowbase + wm * 32 + mi * 16 + j + lhalf * 8;
        const int gcol = colbase + wn * 64 + ni * 16 + l16;
        const float v = acc[mi][ni][j];
        if (MODE == EPI_BF16) {
          ((bf16_t*)Out)[(size_t)grow * Nfull + gcol] = (bf16_t)v;
        } else if (MODE == EPI_RESID) {
          float o = resid[(size_t)grow * Nfull + gcol] + lsv[gcol] * (v + bip[gcol]);
          ((float*)Out)[(size_t)grow * Nfull + gcol] = o;
        } else if (MODE == EPI_GELU_BF16) {
          ((bf16_t*)Out)[(size_t)grow * Nfull + gcol] = (bf16_t)gelu_tanh(v + bip[gcol]);
        } else {  // EPI_MOE_SCATTER
          int tk = tok_list[grow];
          if (tk >= 0)
            atomicAdd(&moe_acc[(size_t)tk * Nfull + gcol],
                      gate_list[grow] * (v + bip[gcol]));
        }
      }
    }
  }
}

// ---------------- Flash attention: one block per (q-tile 64, head, batch) ----------------
__global__ __launch_bounds__(128) void attn_kernel(
    const bf16_t* __restrict__ q, const bf16_t* __restrict__ k,
    const bf16_t* __restrict__ v, bf16_t* __restrict__ aout) {
  const int qt = blockIdx.x, h = blockIdx.y, b = blockIdx.z;
  __shared__ __align__(16) bf16_t qs[64][72];
  __shared__ __align__(16) bf16_t ksm[64][72];
  __shared__ __align__(16) bf16_t vt[64][72];   // [dh][key]
  __shared__ __align__(16) bf16_t ps[64][72];
  __shared__ float sp[64][68];
  __shared__ float rowm[64], rowl[64], rowsc[64];
  const int tid = threadIdx.x;
  const int lane = tid & 31, w = tid >> 5;
  const int l16 = lane & 15, lhalf = lane >> 4;

  // Q tile via async DMA
#pragma unroll
  for (int it = 0; it < 4; ++it) {
    int idx = (tid + it * 128) * 8;
    int r = idx >> 6, c = idx & 63;
    async_b128(q + (size_t)(b * kSeq + qt * 64 + r) * kD + h * kHd + c, &qs[r][c]);
  }
  if (tid < 64) { rowm[tid] = -1e30f; rowl[tid] = 0.f; }
  v8f oacc[4] = {};
  wait_async0();
  __syncthreads();

  for (int kt = 0; kt < kSeq / 64; ++kt) {
    __syncthreads();
    // K tile via async DMA; V tile through VGPRs (needs transpose)
#pragma unroll
    for (int it = 0; it < 4; ++it) {
      int idx = (tid + it * 128) * 8;
      int r = idx >> 6, c = idx & 63;
      async_b128(k + (size_t)(b * kSeq + kt * 64 + r) * kD + h * kHd + c, &ksm[r][c]);
      v8bf tv = *(const v8bf*)(v + (size_t)(b * kSeq + kt * 64 + r) * kD + h * kHd + c);
#pragma unroll
      for (int i = 0; i < 8; ++i) vt[c + i][r] = tv[i];
    }
    wait_async0();
    __syncthreads();
    // S = Q * K^T   (K tile [key][dh] is exactly the B^T layout needed)
    v8f sacc[4] = {};
#pragma unroll
    for (int ks = 0; ks < 64; ks += 32) {
      v16bf aq = frag_a(&qs[w * 16 + l16][ks], lhalf);
#pragma unroll
      for (int ni = 0; ni < 4; ++ni) {
        v16bf bk = frag_b(&ksm[ni * 16 + l16][ks], lhalf);
        sacc[ni] = wmma_bf16(aq, bk, sacc[ni]);
      }
    }
#pragma unroll
    for (int ni = 0; ni < 4; ++ni)
#pragma unroll
      for (int j = 0; j < 8; ++j)
        sp[w * 16 + j + lhalf * 8][ni * 16 + l16] = sacc[ni][j] * kScale;
    __syncthreads();
    if (tid < 64) {  // online softmax, one thread per row
      float mprev = rowm[tid], tm = mprev;
      for (int c = 0; c < 64; ++c) tm = fmaxf(tm, sp[tid][c]);
      float rs = __expf(mprev - tm);
      float l = rowl[tid] * rs;
      for (int c = 0; c < 64; ++c) {
        float p = __expf(sp[tid][c] - tm);
        l += p;
        ps[tid][c] = (bf16_t)p;
      }
      rowm[tid] = tm; rowl[tid] = l; rowsc[tid] = rs;
    }
    __syncthreads();
#pragma unroll
    for (int j = 0; j < 8; ++j) {
      float f = rowsc[w * 16 + j + lhalf * 8];
#pragma unroll
      for (int ni = 0; ni < 4; ++ni) oacc[ni][j] *= f;
    }
#pragma unroll
    for (int ks = 0; ks < 64; ks += 32) {
      v16bf ap = frag_a(&ps[w * 16 + l16][ks], lhalf);
#pragma unroll
      for (int ni = 0; ni < 4; ++ni) {
        v16bf bv = frag_b(&vt[ni * 16 + l16][ks], lhalf);
        oacc[ni] = wmma_bf16(ap, bv, oacc[ni]);
      }
    }
  }
  __syncthreads();
#pragma unroll
  for (int j = 0; j < 8; ++j) {
    float inv = 1.f / rowl[w * 16 + j + lhalf * 8];
#pragma unroll
    for (int ni = 0; ni < 4; ++ni) {
      size_t row = (size_t)(b * kSeq + qt * 64 + w * 16 + j + lhalf * 8);
      aout[row * kD + h * kHd + ni * 16 + l16] = (bf16_t)(oacc[ni][j] * inv);
    }
  }
}

// ---------------- Router: one wave per token ----------------
__global__ __launch_bounds__(32) void router_kernel(
    const bf16_t* __restrict__ xm, const float* __restrict__ rw,
    int* __restrict__ topi, float* __restrict__ topg, int* __restrict__ cnt) {
  const int t = blockIdx.x;
  const int lane = threadIdx.x;
  float acc[kE] = {};
  for (int d = lane; d < kD; d += 32) {
    float xv = (float)xm[(size_t)t * kD + d];
    const float* wr = rw + (size_t)d * kE;
#pragma unroll
    for (int e2 = 0; e2 < kE; ++e2) acc[e2] += xv * wr[e2];
  }
  __shared__ float red[32][kE];
#pragma unroll
  for (int e2 = 0; e2 < kE; ++e2) red[lane][e2] = acc[e2];
  __syncthreads();
  if (lane == 0) {
    float lg[kE];
    for (int e2 = 0; e2 < kE; ++e2) {
      float s = 0.f;
      for (int l = 0; l < 32; ++l) s += red[l][e2];
      lg[e2] = s;
    }
    float mx = lg[0];
    for (int e2 = 1; e2 < kE; ++e2) mx = fmaxf(mx, lg[e2]);
    float p[kE], sum = 0.f;
    for (int e2 = 0; e2 < kE; ++e2) { p[e2] = __expf(lg[e2] - mx); sum += p[e2]; }
    float invs = 1.f / sum;
    for (int e2 = 0; e2 < kE; ++e2) p[e2] *= invs;
    int i0 = 0;
    for (int e2 = 1; e2 < kE; ++e2) if (p[e2] > p[i0]) i0 = e2;
    int i1 = (i0 == 0) ? 1 : 0;
    for (int e2 = 0; e2 < kE; ++e2) if (e2 != i0 && p[e2] > p[i1]) i1 = e2;
    float inv2 = 1.f / (p[i0] + p[i1]);
    topi[t * 2] = i0; topi[t * 2 + 1] = i1;
    topg[t * 2] = p[i0] * inv2; topg[t * 2 + 1] = p[i1] * inv2;
    atomicAdd(&cnt[i0], 1);
    atomicAdd(&cnt[i1], 1);
  }
}

__global__ void init_kernel(float* moe, int* tok, int* cnt, int* fill, float* zsrc) {
  int i = blockIdx.x * blockDim.x + threadIdx.x;
  if (i < kT * kD) moe[i] = 0.f;
  if (i < kPadRows) tok[i] = -1;
  if (i < kE) { cnt[i] = 0; fill[i] = 0; }
  if (i < 64) zsrc[i] = 0.f;   // 256B zero page for masked gather rows
}

__global__ void scan_kernel(const int* cnt, int* off) {
  if (threadIdx.x == 0 && blockIdx.x == 0) {
    int o2 = 0;
    for (int e2 = 0; e2 < kE; ++e2) {
      off[e2] = o2;
      o2 += ((cnt[e2] + BM - 1) / BM) * BM;   // 128-aligned segments
    }
    off[kE] = o2;
  }
}

__global__ void scatter_kernel(const int* __restrict__ topi, const float* __restrict__ topg,
                               const int* __restrict__ soff, int* __restrict__ fill,
                               int* __restrict__ tok, float* __restrict__ gate) {
  int t = blockIdx.x * blockDim.x + threadIdx.x;
  if (t >= kT) return;
  for (int k2 = 0; k2 < 2; ++k2) {
    int e2 = topi[t * 2 + k2];
    int pos = atomicAdd(&fill[e2], 1);
    int slot = soff[e2] + pos;
    tok[slot] = t;
    gate[slot] = topg[t * 2 + k2];
  }
}

__global__ void final_kernel(const float* __restrict__ x2, const float* __restrict__ ls2,
                             const float* __restrict__ moe, float* __restrict__ out) {
  int i = blockIdx.x * blockDim.x + threadIdx.x;
  if (i < kT * kD) out[i] = x2[i] + ls2[i & (kD - 1)] * moe[i];
}

// ---------------- host launcher ----------------
extern "C" void kernel_launch(void* const* d_in, const int* in_sizes, int n_in,
                              void* d_out, int out_size, void* d_ws, size_t ws_size,
                              hipStream_t stream) {
  (void)in_sizes; (void)n_in; (void)out_size; (void)ws_size;
  const float* x    = (const float*)d_in[0];
  const float* ln1g = (const float*)d_in[1];
  const float* ln1b = (const float*)d_in[2];
  const float* wq   = (const float*)d_in[3];
  const float* wk   = (const float*)d_in[4];
  const float* wv   = (const float*)d_in[5];
  const float* wo   = (const float*)d_in[6];
  const float* bo   = (const float*)d_in[7];
  const float* ls1  = (const float*)d_in[8];
  const float* ln2g = (const float*)d_in[9];
  const float* ln2b = (const float*)d_in[10];
  const float* rw   = (const float*)d_in[11];
  const float* fc1w = (const float*)d_in[12];
  const float* fc1b = (const float*)d_in[13];
  const float* fc2w = (const float*)d_in[14];
  const float* fc2b = (const float*)d_in[15];
  const float* ls2  = (const float*)d_in[16];
  float* out = (float*)d_out;

  char* ws = (char*)d_ws;
  size_t o = 0;
  auto take = [&](size_t bytes) {
    o = (o + 255) & ~(size_t)255;
    char* p = ws + o;
    o += bytes;
    return p;
  };
  bf16_t* xn   = (bf16_t*)take((size_t)kT * kD * 2);
  bf16_t* qb   = (bf16_t*)take((size_t)kT * kD * 2);
  bf16_t* kb   = (bf16_t*)take((size_t)kT * kD * 2);
  bf16_t* vb   = (bf16_t*)take((size_t)kT * kD * 2);
  bf16_t* ab   = (bf16_t*)take((size_t)kT * kD * 2);
  float*  x2   = (float*) take((size_t)kT * kD * 4);
  bf16_t* xm   = (bf16_t*)take((size_t)kT * kD * 2);
  bf16_t* hbuf = (bf16_t*)take((size_t)kPadRows * kF * 2);
  float*  moe  = (float*) take((size_t)kT * kD * 4);
  int*    topi = (int*)   take((size_t)kT * 2 * 4);
  float*  topg = (float*) take((size_t)kT * 2 * 4);
  int*    cnt  = (int*)   take(kE * 4);
  int*    fill = (int*)   take(kE * 4);
  int*    soff = (int*)   take((kE + 1) * 4);
  int*    tok  = (int*)   take((size_t)kPadRows * 4);
  float*  gate = (float*) take((size_t)kPadRows * 4);
  float*  zsrc = (float*) take(256);

  // --- attention sublayer ---
  ln_kernel<<<kT, 256, 0, stream>>>(x, ln1g, ln1b, xn);
  const dim3 gproj(kD / BN, kT / BM);  // 8 x 16
  gemm_kernel<EPI_BF16, false, false><<<gproj, 256, 0, stream>>>(
      xn, wq, nullptr, qb, nullptr, nullptr, nullptr, nullptr, nullptr, nullptr,
      nullptr, kD, kD, kD);
  gemm_kernel<EPI_BF16, false, false><<<gproj, 256, 0, stream>>>(
      xn, wk, nullptr, kb, nullptr, nullptr, nullptr, nullptr, nullptr, nullptr,
      nullptr, kD, kD, kD);
  gemm_kernel<EPI_BF16, false, false><<<gproj, 256, 0, stream>>>(
      xn, wv, nullptr, vb, nullptr, nullptr, nullptr, nullptr, nullptr, nullptr,
      nullptr, kD, kD, kD);
  attn_kernel<<<dim3(kSeq / 64, kH, kB), 128, 0, stream>>>(qb, kb, vb, ab);
  gemm_kernel<EPI_RESID, false, false><<<gproj, 256, 0, stream>>>(
      ab, wo, bo, x2, x, ls1, nullptr, nullptr, nullptr, nullptr,
      nullptr, kD, kD, kD);

  // --- MoE sublayer ---
  ln_kernel<<<kT, 256, 0, stream>>>(x2, ln2g, ln2b, xm);
  init_kernel<<<(kT * kD + 255) / 256, 256, 0, stream>>>(moe, tok, cnt, fill, zsrc);
  router_kernel<<<kT, 32, 0, stream>>>(xm, rw, topi, topg, cnt);
  scan_kernel<<<1, 1, 0, stream>>>(cnt, soff);
  scatter_kernel<<<(kT + 255) / 256, 256, 0, stream>>>(topi, topg, soff, fill, tok, gate);
  gemm_kernel<EPI_GELU_BF16, true, true><<<dim3(kF / BN, kPadRows / BM), 256, 0, stream>>>(
      xm, fc1w, fc1b, hbuf, nullptr, nullptr, tok, nullptr, soff, nullptr,
      (const bf16_t*)zsrc, kF, kD, kD);
  gemm_kernel<EPI_MOE_SCATTER, false, true><<<dim3(kD / BN, kPadRows / BM), 256, 0, stream>>>(
      hbuf, fc2w, fc2b, nullptr, nullptr, nullptr, tok, gate, soff, moe,
      nullptr, kD, kF, kF);
  final_kernel<<<(kT * kD + 255) / 256, 256, 0, stream>>>(x2, ls2, moe, out);
}
